// SelfAttention_63376537420539
// MI455X (gfx1250) — matrix operand order repaired and checked by reading, not compiled, not run on hardware
//
#include <hip/hip_runtime.h>
#include <hip/hip_bf16.h>
#include <stdint.h>

// ---------------------------------------------------------------------------
// Self-attention (row-sum normalized) for A[8192,256] fp32.
//   S = A A^T ; out = (S / rowsum(S)) A
// Fused flash-style: never materialize S. bf16 WMMA (16x16x32) with f32 acc.
// rowsum_i = a_i . colsum(A)  (exact identity), applied in a final pass.
// matmul1 computes S^T = K Q^T (Q resident as B operand) so staged S can be
// written with packed b64 stores; each streamed LDS fragment feeds 2 WMMAs.
// ---------------------------------------------------------------------------

#define N_ROWS   8192
#define DIM      256
#define TILE_J   128      // streamed K/V tile rows
#define MTILE    128      // rows of output per block (32 per wave pair)
#define NCHUNK   4        // j-range split across blocks
#define NTILES   (N_ROWS / TILE_J)           // 64
#define TPC      (NTILES / NCHUNK)           // 16 tiles per chunk
#define SMEM_BYTES (65536*2 + 65536*2 + 32768)   // 288 KB

typedef __attribute__((ext_vector_type(16))) __bf16 v16bf;
typedef __attribute__((ext_vector_type(2)))  __bf16 v2bf;
typedef __attribute__((ext_vector_type(8)))  float  v8f;
typedef __attribute__((ext_vector_type(2)))  float  v2f;

union Frag {
    v16bf v;
    uint4 q[2];
};

__device__ inline unsigned bfpack(float a, float b) {
    // vector fptrunc -> single v_cvt_pk_bf16_f32
    v2f f;
    f.x = a;
    f.y = b;
    v2bf h = __builtin_convertvector(f, v2bf);
    return __builtin_bit_cast(unsigned, h);
}

#define WMMA_BF16(a, b, c) \
    __builtin_amdgcn_wmma_f32_16x16x32_bf16(false, (a).v, false, (b).v, (short)0, (c), false, false)

// Load one 128x256 fp32 tile from global, convert to bf16, store row-major
// into bK; if TR, also store transposed (d-major) into bKT with packed b32
// stores (2 rows per store). Each thread covers 2 rows x 4 cols per step.
template <bool TR>
__device__ inline void load_tile(const float* __restrict__ src,
                                 unsigned short* __restrict__ bK,
                                 unsigned short* __restrict__ bKT) {
    const int t  = threadIdx.x;
    const int c0 = (t * 4) & 255;      // 4 fixed columns per thread
    const int rp = (t >> 6) * 2;       // row-pair phase 0,2,4,6
    float4 ra[4], rb[4];

    #pragma unroll
    for (int s = 0; s < 2; ++s) {
        const int r = s * 8 + rp;
        ra[s * 2 + 0] = *(const float4*)&src[(r + 0) * DIM + c0];
        ra[s * 2 + 1] = *(const float4*)&src[(r + 1) * DIM + c0];
    }
    #pragma unroll
    for (int ch = 0; ch < 8; ++ch) {
        float4* cur = (ch & 1) ? rb : ra;
        float4* nxt = (ch & 1) ? ra : rb;
        if (ch < 7) {
            #pragma unroll
            for (int s = 0; s < 2; ++s) {
                const int r = (ch + 1) * 16 + s * 8 + rp;
                nxt[s * 2 + 0] = *(const float4*)&src[(r + 0) * DIM + c0];
                nxt[s * 2 + 1] = *(const float4*)&src[(r + 1) * DIM + c0];
            }
        }
        #pragma unroll
        for (int s = 0; s < 2; ++s) {
            const int r = ch * 16 + s * 8 + rp;
            float4 v0 = cur[s * 2 + 0];
            float4 v1 = cur[s * 2 + 1];
            uint2 row0, row1;
            row0.x = bfpack(v0.x, v0.y);
            row0.y = bfpack(v0.z, v0.w);
            row1.x = bfpack(v1.x, v1.y);
            row1.y = bfpack(v1.z, v1.w);
            *(uint2*)&bK[(r + 0) * DIM + c0] = row0;   // ds_store_b64
            *(uint2*)&bK[(r + 1) * DIM + c0] = row1;
            if (TR) {                                   // b32, 2 rows packed
                *(unsigned*)&bKT[(c0 + 0) * TILE_J + r] = bfpack(v0.x, v1.x);
                *(unsigned*)&bKT[(c0 + 1) * TILE_J + r] = bfpack(v0.y, v1.y);
                *(unsigned*)&bKT[(c0 + 2) * TILE_J + r] = bfpack(v0.z, v1.z);
                *(unsigned*)&bKT[(c0 + 3) * TILE_J + r] = bfpack(v0.w, v1.w);
            }
        }
    }
}

// ---------------------------------------------------------------------------
// Kernel A: colsum[d] = sum_i A[i][d]   (ws must be zeroed first)
// ---------------------------------------------------------------------------
__global__ __launch_bounds__(256) void colsum_kernel(const float* __restrict__ A,
                                                     float* __restrict__ cs) {
    const int t = threadIdx.x;                 // column
    const float* p = A + (size_t)blockIdx.x * 128 * DIM + t;
    float s = 0.f;
    #pragma unroll 8
    for (int r = 0; r < 128; ++r) s += p[(size_t)r * DIM];
    atomicAdd(&cs[t], s);
}

// ---------------------------------------------------------------------------
// Kernel B: unnormalized O += (Q K^T) V over this block's j-chunk, atomically
// accumulated into d_out (zero-initialized by the launcher).
// Wave pair owns rows pair*32..pair*32+31; `half` splits the jt range
// (matmul1) and the d range (matmul2) so each streamed fragment is reused
// by 2 WMMAs against two resident fragments.
// ---------------------------------------------------------------------------
__global__ __launch_bounds__(256) void attn_kernel(const float* __restrict__ A,
                                                   float* __restrict__ out) {
    extern __shared__ char smem[];
    unsigned short* bK0 = (unsigned short*)(smem);            // [128][256] bf16
    unsigned short* bK1 = (unsigned short*)(smem + 65536);
    unsigned short* bT0 = (unsigned short*)(smem + 131072);   // [256][128] bf16
    unsigned short* bT1 = (unsigned short*)(smem + 196608);
    unsigned short* sS  = (unsigned short*)(smem + 262144);   // [4][32 i][128 j] bf16

    const int t    = threadIdx.x;
    const int w    = t >> 5;
    const int l    = t & 31;
    const int lh   = l >> 4;        // lane half
    const int ll   = l & 15;
    const int pair = w >> 1;        // 0..3 -> rows pair*32..pair*32+31
    const int half = w & 1;         // splits jt (mm1) and d (mm2) ranges

    const int mblk  = (int)blockIdx.x >> 2;   // 0..63
    const int chunk = (int)blockIdx.x & 3;    // 0..3
    const int j0    = chunk * TPC;            // first tile of this chunk

    // ---- Q tile (row-major only) into bK1, build resident Q^T B-frags ----
    load_tile<false>(A + (size_t)mblk * MTILE * DIM, bK1, nullptr);
    __syncthreads();

    Frag qB[2][8];                  // 2 row groups x 8 k-chunks (B operand: Q^T)
    #pragma unroll
    for (int g = 0; g < 2; ++g) {
        // B[k=d][n=i] = Q[i][d]; B 32x16: lo=K0..7, hi=K16..23 (+8*lh)
        const unsigned short* qb = bK1 + (pair * 32 + g * 16 + ll) * DIM + 8 * lh;
        #pragma unroll
        for (int kb = 0; kb < 8; ++kb) {
            qB[g][kb].q[0] = *(const uint4*)&qb[kb * 32];
            qB[g][kb].q[1] = *(const uint4*)&qb[kb * 32 + 16];
        }
    }

    // ---- first streamed tile into buffer 0 (both copies) ----
    load_tile<true>(A + (size_t)j0 * TILE_J * DIM, bK0, bT0);
    __syncthreads();

    v8f acc[2][8];                  // 2 row groups x 8 d-tiles (this half)
    #pragma unroll
    for (int g = 0; g < 2; ++g)
        #pragma unroll
        for (int i = 0; i < 8; ++i)
            acc[g][i] = v8f{0.f, 0.f, 0.f, 0.f, 0.f, 0.f, 0.f, 0.f};

    unsigned short* sSp = sS + pair * (32 * TILE_J);

    for (int it = 0; it < TPC; ++it) {
        const unsigned short* bK = (it & 1) ? bK1 : bK0;
        const unsigned short* bT = (it & 1) ? bT1 : bT0;

        // ---- prefetch next tile's cachelines while WMMAs run ----
        if (it + 1 < TPC) {
            const char* np = (const char*)(A + (size_t)(j0 + it + 1) * TILE_J * DIM);
            #pragma unroll
            for (int pf = 0; pf < 4; ++pf)
                __builtin_prefetch(np + ((size_t)t + pf * 256) * 128, 0, 1);
        }

        // ---- matmul1: S^T(128x32) = K(128x256) * Q^T; this wave: 4 j-tiles --
        #pragma unroll
        for (int jti = 0; jti < 4; ++jti) {
            const int jt = half * 4 + jti;
            // A[m=j][k=d] from row-major K tile; A 16x32: lo=K0..7|8..15 (+8*lh)
            const unsigned short* ar = bK + (jt * 16 + ll) * DIM + 8 * lh;
            v8f c0 = v8f{0.f, 0.f, 0.f, 0.f, 0.f, 0.f, 0.f, 0.f};
            v8f c1 = v8f{0.f, 0.f, 0.f, 0.f, 0.f, 0.f, 0.f, 0.f};
            Frag a0, a1;            // double-buffered streamed A fragments
            a0.q[0] = *(const uint4*)&ar[0];
            a0.q[1] = *(const uint4*)&ar[16];
            #pragma unroll
            for (int kb = 0; kb < 8; ++kb) {
                Frag& curf = (kb & 1) ? a1 : a0;
                Frag& nxtf = (kb & 1) ? a0 : a1;
                if (kb < 7) {
                    nxtf.q[0] = *(const uint4*)&ar[(kb + 1) * 32];
                    nxtf.q[1] = *(const uint4*)&ar[(kb + 1) * 32 + 16];
                }
                c0 = WMMA_BF16(curf, qB[0][kb], c0);
                c1 = WMMA_BF16(curf, qB[1][kb], c1);
            }
            // stage S^T tile: D rows = j (contiguous in S[i][j]) -> b64 stores
            unsigned short* s0 = sSp + ll * TILE_J + jt * 16 + 8 * lh;        // g=0
            unsigned short* s1 = sSp + (16 + ll) * TILE_J + jt * 16 + 8 * lh; // g=1
            uint2 u;
            u.x = bfpack(c0[0], c0[1]); u.y = bfpack(c0[2], c0[3]);
            *(uint2*)&s0[0] = u;
            u.x = bfpack(c0[4], c0[5]); u.y = bfpack(c0[6], c0[7]);
            *(uint2*)&s0[4] = u;
            u.x = bfpack(c1[0], c1[1]); u.y = bfpack(c1[2], c1[3]);
            *(uint2*)&s1[0] = u;
            u.x = bfpack(c1[4], c1[5]); u.y = bfpack(c1[6], c1[7]);
            *(uint2*)&s1[4] = u;
        }
        __syncthreads();   // S(32x128) shared across the wave pair

        // ---- matmul2: O(32x128-half) += S(32x128) * V; 8 d-tiles/wave ----
        #pragma unroll
        for (int kc = 0; kc < 4; ++kc) {
            Frag sa0, sa1;          // S as A operands (row-major, K=j contiguous)
            const unsigned short* sb0 = sSp + ll * TILE_J + kc * 32 + 8 * lh;
            const unsigned short* sb1 = sb0 + 16 * TILE_J;
            sa0.q[0] = *(const uint4*)&sb0[0];
            sa0.q[1] = *(const uint4*)&sb0[16];
            sa1.q[0] = *(const uint4*)&sb1[0];
            sa1.q[1] = *(const uint4*)&sb1[16];
            const unsigned short* tbb =
                bT + ((half * 8) * 16 + ll) * TILE_J + kc * 32 + 8 * lh;
            Frag b0, b1;
            b0.q[0] = *(const uint4*)&tbb[0];
            b0.q[1] = *(const uint4*)&tbb[16];
            #pragma unroll
            for (int db = 0; db < 8; ++db) {
                Frag& curf = (db & 1) ? b1 : b0;
                Frag& nxtf = (db & 1) ? b0 : b1;
                if (db < 7) {
                    const unsigned short* tb = tbb + (db + 1) * (16 * TILE_J);
                    nxtf.q[0] = *(const uint4*)&tb[0];
                    nxtf.q[1] = *(const uint4*)&tb[16];
                }
                acc[0][db] = WMMA_BF16(sa0, curf, acc[0][db]);
                acc[1][db] = WMMA_BF16(sa1, curf, acc[1][db]);
            }
        }

        // ---- load next tile into the other buffer ----
        if (it + 1 < TPC) {
            unsigned short* nK = (it & 1) ? bK0 : bK1;
            unsigned short* nT = (it & 1) ? bT0 : bT1;
            load_tile<true>(A + (size_t)(j0 + it + 1) * TILE_J * DIM, nK, nT);
        }
        __syncthreads();
    }

    // ---- accumulate unnormalized O into global ----
    #pragma unroll
    for (int g = 0; g < 2; ++g) {
        float* ob = out + (size_t)(mblk * MTILE + pair * 32 + g * 16 + 8 * lh) * DIM
                        + half * 128 + ll;
        #pragma unroll
        for (int db = 0; db < 8; ++db) {
            #pragma unroll
            for (int r = 0; r < 8; ++r)
                atomicAdd(&ob[(size_t)r * DIM + db * 16], acc[g][db][r]);
        }
    }
}

// ---------------------------------------------------------------------------
// Kernel C: out[i][:] /= (a_i . colsum)   -- one wave per row
// ---------------------------------------------------------------------------
__global__ __launch_bounds__(256) void normalize_kernel(const float* __restrict__ A,
                                                        const float* __restrict__ cs,
                                                        float* __restrict__ out) {
    const int w = threadIdx.x >> 5;
    const int l = threadIdx.x & 31;
    const int row = (int)blockIdx.x * 8 + w;
    const float* a = A + (size_t)row * DIM;
    float s = 0.f;
    #pragma unroll
    for (int k = 0; k < 8; ++k) s += a[l + 32 * k] * cs[l + 32 * k];
    #pragma unroll
    for (int off = 16; off; off >>= 1) s += __shfl_xor(s, off, 32);
    const float inv = 1.0f / s;
    float* o = out + (size_t)row * DIM;
    #pragma unroll
    for (int k = 0; k < 8; ++k) o[l + 32 * k] *= inv;
}

// ---------------------------------------------------------------------------
extern "C" void kernel_launch(void* const* d_in, const int* in_sizes, int n_in,
                              void* d_out, int out_size, void* d_ws, size_t ws_size,
                              hipStream_t stream) {
    (void)in_sizes; (void)n_in; (void)out_size; (void)ws_size;
    const float* A   = (const float*)d_in[0];
    float*       out = (float*)d_out;
    float*       cs  = (float*)d_ws;          // 256 floats of scratch

    hipMemsetAsync(d_out, 0, sizeof(float) * (size_t)N_ROWS * DIM, stream);
    hipMemsetAsync(d_ws, 0, sizeof(float) * DIM, stream);

    colsum_kernel<<<NTILES, 256, 0, stream>>>(A, cs);
    attn_kernel<<<NTILES * NCHUNK, 256, SMEM_BYTES, stream>>>(A, out);
    normalize_kernel<<<N_ROWS / 8, 256, 0, stream>>>(A, cs, out);
}
